// MemoryGate_91096256348183
// MI455X (gfx1250) — matrix operand order repaired and checked by reading, not compiled, not run on hardware
//
#include <hip/hip_runtime.h>

// ---------------------------------------------------------------------------
// MemoryGate for MI455X (gfx1250, wave32, WMMA 16x16x32 f16 -> f32)
// One 128-thread block (4 wave32) per (b,n); every GEMM (QKV proj, QK^T, PV,
// input-memory energy, memory readout) runs on v_wmma_f32_16x16x32_f16.
// Wave id is readfirstlane'd and all tile indices are compile-time so EXEC
// stays all-ones at every WMMA site (ISA requirement).
// ---------------------------------------------------------------------------

typedef __attribute__((ext_vector_type(16))) _Float16 v16h;
typedef __attribute__((ext_vector_type(8)))  _Float16 v8h;
typedef __attribute__((ext_vector_type(4)))  _Float16 v4h;
typedef __attribute__((ext_vector_type(8)))  float    v8f;

#define WMMA_F16(a, b, c) \
  __builtin_amdgcn_wmma_f32_16x16x32_f16(false, (a), false, (b), (short)0, (c), false, false)

constexpr int Bc = 4, Nc = 2048, Tc = 64, HIDc = 64, Mc = 32, KMEM = 20;
constexpr float EPSc = 1e-8f;

// LDS strides (elements). f16 strides are multiples of 8 -> 16B-aligned rows.
constexpr int LDH = 72;   // sH, sP, sVt, sWt rows
constexpr int LDQ = 40;   // sQ, sK, sBankH, sBankT rows
constexpr int LDE = 65;   // sE (f32)
constexpr int LDA = 33;   // sAtt, sMem (f32)

// A fragment: 16x32 f16 row-major [rows][ld].
// half j of lane L holds A[row0+(L&15)][k0 + (j>>3)*16 + ((L>>4)<<3) + (j&7)]
__device__ __forceinline__ v16h load_a_frag(const _Float16* base, int row0, int k0,
                                            int ld, int lane) {
  int r  = row0 + (lane & 15);
  int kb = k0 + ((lane >> 4) << 3);
  const _Float16* p = base + r * ld + kb;
  v8h lo = *(const v8h*)p;          // K = kb .. kb+7
  v8h hi = *(const v8h*)(p + 16);   // K = kb+16 .. kb+23
  v16h a;
#pragma unroll
  for (int j = 0; j < 8; ++j) { a[j] = lo[j]; a[j + 8] = hi[j]; }
  return a;
}

// B fragment: 32x16 f16 from transposed storage baseT[n][ld] = B[k][n].
// half j of lane L holds B[k0 + (j>>3)*16 + ((L>>4)<<3) + (j&7)][n0 + (L&15)]
__device__ __forceinline__ v16h load_b_frag(const _Float16* baseT, int n0, int k0,
                                            int ld, int lane) {
  int n  = n0 + (lane & 15);
  int kb = k0 + ((lane >> 4) << 3);
  const _Float16* p = baseT + n * ld + kb;
  v8h lo = *(const v8h*)p;
  v8h hi = *(const v8h*)(p + 16);
  v16h b;
#pragma unroll
  for (int j = 0; j < 8; ++j) { b[j] = lo[j]; b[j + 8] = hi[j]; }
  return b;
}

__global__ __launch_bounds__(128)
void memgate_kernel(const float* __restrict__ input,
                    const float* __restrict__ h0,
                    const float* __restrict__ h1,
                    const float* __restrict__ h2,
                    const float* __restrict__ memory,
                    const float* __restrict__ hid_query,
                    const float* __restrict__ key_w,
                    const float* __restrict__ value_w,
                    const float* __restrict__ input_query,
                    float* __restrict__ out) {
  // ---- LDS unions (static total ~61 KB) ----
  __shared__ __align__(16) unsigned char smemA[Tc * LDE * 4];      // sE(f32) / sH(f16)
  __shared__ __align__(16) unsigned char smemB[3 * Mc * LDH * 2];  // sWt / sP
  __shared__ __align__(16) unsigned char smemC[Tc * LDA * 4];      // sAtt / {banks,sIn}
  __shared__ __align__(16) _Float16 sQ[Tc * LDQ];
  __shared__ __align__(16) _Float16 sK[Tc * LDQ];
  __shared__ __align__(16) _Float16 sVt[Mc * LDH];                 // V transposed [m][s]
  __shared__ float sMem[Tc * LDA];
  __shared__ float sInQ[2 * Mc];
  __shared__ float sNm[Tc];

  _Float16* sH     = (_Float16*)smemA;          // h tile f16 (until QKV done)
  float*    sE     = (float*)smemA;             // scores (after QKV)
  _Float16* sWt    = (_Float16*)smemB;          // weights (until QKV done)
  _Float16* sP     = (_Float16*)smemB;          // probs (after softmax)
  float*    sAtt   = (float*)smemC;             // head phase only
  _Float16* sBankH = (_Float16*)smemC;          // [32][40] memory row-major (mem phase)
  _Float16* sBankT = (_Float16*)(smemC + 2560); // [32][40] memory transposed (mem phase)
  float*    sIn    = (float*)(smemC + 5120);    // [128] this block's input rows

  const int tid  = threadIdx.x;
  const int lane = tid & 31;
  const int wave = __builtin_amdgcn_readfirstlane(tid >> 5);  // provably uniform
  const int bn   = blockIdx.x;
  const int mm0L = ((lane >> 4) << 3);   // D-tile: row base owned by this lane
  const int nnL  = (lane & 15);          // D-tile: column owned by this lane

  // ================= memory-bank path (head independent) =================
  // zero-pad bank tiles (slots 20..31 must be 0 / ignorable)
  for (int e = tid; e < Mc * LDQ; e += 128) { sBankH[e] = (_Float16)0.f; sBankT[e] = (_Float16)0.f; }
  sIn[tid] = input[(size_t)bn * Tc * 2 + tid];          // 128 = T*IN_DIM floats
  if (tid < 2 * Mc) sInQ[tid] = input_query[tid];
  __syncthreads();

  // fill f16 bank copies + q = input @ input_query (K=2, scalar is optimal)
  for (int e = tid; e < KMEM * Mc; e += 128) {
    float v = memory[e];
    int slot = e >> 5, m = e & 31;
    sBankH[slot * LDQ + m]  = (_Float16)v;
    sBankT[m * LDQ + slot]  = (_Float16)v;
  }
  for (int e = tid; e < Tc * Mc; e += 128) {
    int t = e >> 5, m = e & 31;
    float v = sIn[2 * t] * sInQ[m] + sIn[2 * t + 1] * sInQ[Mc + m];
    sQ[t * LDQ + m] = (_Float16)v;
  }
  __syncthreads();

  // energy = q @ memory^T  (64x32 @ 32x32pad), wave -> row tile
  {
    const int rt = wave;
    v16h a = load_a_frag(sQ, rt * 16, 0, LDQ, lane);
#pragma unroll
    for (int ct = 0; ct < 2; ++ct) {   // cols 0..31 (only 0..19 meaningful)
      v16h b = load_b_frag(sBankH, ct * 16, 0, LDQ, lane);  // B[k=m][n=slot]=mem[slot][m]
      v8f acc = {};
      acc = WMMA_F16(a, b, acc);
      const int mm = rt * 16 + mm0L, nn = ct * 16 + nnL;
#pragma unroll
      for (int r = 0; r < 8; ++r) sE[(mm + r) * LDE + nn] = acc[r];
    }
  }
  __syncthreads();

  // softmax over 20 slots, write P padded to 32 (cols 20..31 = 0)
  if (tid < Tc) {
    float* row = sE + tid * LDE;
    float mx = -3.4e38f;
#pragma unroll
    for (int s = 0; s < KMEM; ++s) mx = fmaxf(mx, row[s]);
    float sum = 0.f;
#pragma unroll
    for (int s = 0; s < KMEM; ++s) { float v = __expf(row[s] - mx); row[s] = v; sum += v; }
    const float inv = 1.0f / sum;
    _Float16* prow = sP + tid * LDH;
#pragma unroll
    for (int s = 0; s < KMEM; ++s) prow[s] = (_Float16)(row[s] * inv);
#pragma unroll
    for (int s = KMEM; s < Mc; ++s) prow[s] = (_Float16)0.f;
  }
  __syncthreads();

  // memories = P @ memory  (64x32pad @ 32x32), wave -> row tile
  {
    const int rt = wave;
    v16h a = load_a_frag(sP, rt * 16, 0, LDH, lane);
#pragma unroll
    for (int ct = 0; ct < 2; ++ct) {
      v16h b = load_b_frag(sBankT, ct * 16, 0, LDQ, lane);  // B[k=slot][n=m]=mem[slot][m]
      v8f acc = {};
      acc = WMMA_F16(a, b, acc);
      const int mm = rt * 16 + mm0L, nn = ct * 16 + nnL;
#pragma unroll
      for (int r = 0; r < 8; ++r) sMem[(mm + r) * LDA + nn] = acc[r];
    }
  }
  __syncthreads();

  if (tid < Tc) {
    float n2 = 0.f;
#pragma unroll
    for (int m = 0; m < Mc; ++m) { float v = sMem[tid * LDA + m]; n2 += v * v; }
    sNm[tid] = fmaxf(sqrtf(n2), EPSc);
  }
  __syncthreads();

  // ========================= per-head attention =========================
  for (int head = 0; head < 3; ++head) {
    const float* hp = (head == 0) ? h0 : (head == 1) ? h1 : h2;

    // h tile [64][64] f32 -> f16 LDS
    const float4* hsrc = (const float4*)(hp + (size_t)bn * (Tc * HIDc));
    for (int v = tid; v < (Tc * HIDc) / 4; v += 128) {
      float4 x = hsrc[v];
      int e = v * 4, t = e >> 6, c = e & 63;
      v4h pk = { (_Float16)x.x, (_Float16)x.y, (_Float16)x.z, (_Float16)x.w };
      *(v4h*)(sH + t * LDH + c) = pk;
    }
    // weights transposed: sWt[mat][m][hid] = W[hid][m]
    const float* w0 = hid_query + head * (HIDc * Mc);
    const float* w1 = key_w     + head * (HIDc * Mc);
    const float* w2 = value_w   + head * (HIDc * Mc);
    for (int e = tid; e < HIDc * Mc; e += 128) {
      int hid = e >> 5, m = e & 31;
      sWt[(0 * Mc + m) * LDH + hid] = (_Float16)w0[e];
      sWt[(1 * Mc + m) * LDH + hid] = (_Float16)w1[e];
      sWt[(2 * Mc + m) * LDH + hid] = (_Float16)w2[e];
    }
    __syncthreads();  // (a)

    // prefetch next head's h tile (128 lanes x 128B = 16KB) under this head's math
    if (head < 2) {
      const float* hn = (head == 0) ? h1 : h2;
      __builtin_prefetch((const char*)(hn + (size_t)bn * (Tc * HIDc)) + tid * 128, 0, 1);
    }

    // ---- Q,K,V = h @ W : wave owns row tile rt=wave; mat/ct compile-time ----
    {
      const int rt = wave;
      v8f acc[3][2];
#pragma unroll
      for (int mat = 0; mat < 3; ++mat)
#pragma unroll
        for (int ct = 0; ct < 2; ++ct) acc[mat][ct] = (v8f){};
#pragma unroll
      for (int kk = 0; kk < HIDc; kk += 32) {
        v16h a = load_a_frag(sH, rt * 16, kk, LDH, lane);
#pragma unroll
        for (int mat = 0; mat < 3; ++mat) {
#pragma unroll
          for (int ct = 0; ct < 2; ++ct) {
            v16h b = load_b_frag(sWt + mat * Mc * LDH, ct * 16, kk, LDH, lane);
            acc[mat][ct] = WMMA_F16(a, b, acc[mat][ct]);
          }
        }
      }
      const int mm = rt * 16 + mm0L;
#pragma unroll
      for (int ct = 0; ct < 2; ++ct) {
        const int nn = ct * 16 + nnL;
#pragma unroll
        for (int r = 0; r < 8; ++r) {
          sQ[(mm + r) * LDQ + nn] = (_Float16)acc[0][ct][r];
          sK[(mm + r) * LDQ + nn] = (_Float16)acc[1][ct][r];
        }
        v8h vp;   // V transposed: contiguous 8 halves -> ds_store_b128
#pragma unroll
        for (int r = 0; r < 8; ++r) vp[r] = (_Float16)acc[2][ct][r];
        *(v8h*)(sVt + nn * LDH + mm) = vp;
      }
    }
    __syncthreads();  // (b)

    // ---- E = Q K^T : wave owns row tile, K=32 ----
    {
      const int rt = wave;
      v16h a = load_a_frag(sQ, rt * 16, 0, LDQ, lane);
#pragma unroll
      for (int ct = 0; ct < 4; ++ct) {
        v16h b = load_b_frag(sK, ct * 16, 0, LDQ, lane);  // B[k=m][n=s]=K[s][m]
        v8f acc = {};
        acc = WMMA_F16(a, b, acc);
        const int mm = rt * 16 + mm0L, nn = ct * 16 + nnL;
#pragma unroll
        for (int r = 0; r < 8; ++r) sE[(mm + r) * LDE + nn] = acc[r];
      }
    }
    __syncthreads();  // (c)

    // ---- row softmax over s=0..63, write P f16 ----
    if (tid < Tc) {
      float* row = sE + tid * LDE;
      float mx = -3.4e38f;
#pragma unroll 8
      for (int s = 0; s < Tc; ++s) mx = fmaxf(mx, row[s]);
      float sum = 0.f;
#pragma unroll 8
      for (int s = 0; s < Tc; ++s) { float v = __expf(row[s] - mx); row[s] = v; sum += v; }
      const float inv = 1.0f / sum;
      _Float16* prow = sP + tid * LDH;
#pragma unroll 8
      for (int s = 0; s < Tc; ++s) prow[s] = (_Float16)(row[s] * inv);
    }
    __syncthreads();  // (d)

    // ---- att = P @ V : wave owns row tile, ct compile-time, K=64 ----
    {
      const int rt = wave;
      v8f acc[2];
      acc[0] = (v8f){}; acc[1] = (v8f){};
#pragma unroll
      for (int kk = 0; kk < Tc; kk += 32) {
        v16h a = load_a_frag(sP, rt * 16, kk, LDH, lane);
#pragma unroll
        for (int ct = 0; ct < 2; ++ct) {
          v16h b = load_b_frag(sVt, ct * 16, kk, LDH, lane);  // B[k=s][n=m]=V[s][m]
          acc[ct] = WMMA_F16(a, b, acc[ct]);
        }
      }
      const int mm = rt * 16 + mm0L;
#pragma unroll
      for (int ct = 0; ct < 2; ++ct) {
        const int nn = ct * 16 + nnL;
#pragma unroll
        for (int r = 0; r < 8; ++r) sAtt[(mm + r) * LDA + nn] = acc[ct][r];
      }
    }
    __syncthreads();  // (e)

    // ---- cosine similarity for this head ----
    if (tid < Tc) {
      float dot = 0.f, na2 = 0.f;
#pragma unroll
      for (int m = 0; m < Mc; ++m) {
        const float am = sAtt[tid * LDA + m];
        dot += am * sMem[tid * LDA + m];
        na2 += am * am;
      }
      const float na = fmaxf(sqrtf(na2), EPSc);
      out[((size_t)bn * Tc + tid) * 3 + head] = dot / (sNm[tid] * na);
    }
    __syncthreads();  // (f) before next head reuses LDS
  }
}

extern "C" void kernel_launch(void* const* d_in, const int* in_sizes, int n_in,
                              void* d_out, int out_size, void* d_ws, size_t ws_size,
                              hipStream_t stream) {
  (void)in_sizes; (void)n_in; (void)out_size; (void)d_ws; (void)ws_size;
  const float* input       = (const float*)d_in[0];
  const float* h0          = (const float*)d_in[1];
  const float* h1          = (const float*)d_in[2];
  const float* h2          = (const float*)d_in[3];
  const float* memory      = (const float*)d_in[4];
  const float* hid_query   = (const float*)d_in[5];
  const float* key_w       = (const float*)d_in[6];
  const float* value_w     = (const float*)d_in[7];
  const float* input_query = (const float*)d_in[8];
  float* out = (float*)d_out;

  dim3 grid(Bc * Nc);   // 8192 workgroups, one per (b,n)
  dim3 block(128);      // 4 wave32s
  memgate_kernel<<<grid, block, 0, stream>>>(input, h0, h1, h2, memory,
                                             hid_query, key_w, value_w,
                                             input_query, out);
}
